// Encoder_91027536871674
// MI455X (gfx1250) — matrix operand order repaired and checked by reading, not compiled, not run on hardware
//
#include <hip/hip_runtime.h>
#include <stdint.h>

typedef __bf16 bf16_t;
typedef __attribute__((ext_vector_type(16))) __bf16 v16bf;
typedef __attribute__((ext_vector_type(8)))  float  v8f;

#define B_   2
#define C_   512
#define FC_  2048
#define H_   8
#define DH_  64
#define TX_  1024
#define TT_  2048
#define L_   6

__device__ __forceinline__ bf16_t f2bf(float f){
  union { float f; uint32_t u; } a; a.f = f;
  uint32_t r = a.u + 0x7FFFu + ((a.u >> 16) & 1u);
  union { uint16_t u; bf16_t b; } o; o.u = (uint16_t)(r >> 16);
  return o.b;
}
__device__ __forceinline__ float bf2f(bf16_t b){
  union { uint16_t u; bf16_t b; } i; i.b = b;
  union { uint32_t u; float f; } o; o.u = ((uint32_t)i.u) << 16;
  return o.f;
}

// ---------------------------------------------------------------- GEMM ------
// D = A(MxK) * BT(NxK)^T.  Both operands stored k-contiguous ("BT[n][k]").
struct GemmP {
  const bf16_t* A; const bf16_t* B;     // B is the transposed operand BT[N][K]
  long long batchA, batchB;
  int lda, ldb, K;                      // ldb = row stride of BT (along k)
  const float* bias;
  const float* resid; long long batchRes; int ldres;
  float* outF; bf16_t* outH; long long batchOut; int ldout;
  const float* mask; long long batchMask;
  float scale;
  int epilog;
};
enum { E_PLAIN_F = 0, E_PLAIN_H, E_QSCAT, E_SCORES, E_PVSCAT, E_OPROJ, E_GELU_T, E_CONV2 };

// Block tile 128x64, 8 waves as 4(M)x2(N); wave tile 32x32 -> 4 WMMA per K-step.
// LDS double-buffered; tiles staged with GLOBAL_LOAD_ASYNC_TO_LDS_B128 (ASYNCcnt).
__launch_bounds__(256)
__global__ void gemm_bf16_wmma(GemmP p){
  __shared__ alignas(16) bf16_t As[2][128][40];
  __shared__ alignas(16) bf16_t Bs[2][64][40];
  const int tid  = threadIdx.x;
  const int wave = tid >> 5, lane = tid & 31;
  const int half = lane >> 4, l16 = lane & 15;
  const int wm = (wave & 3) * 32;   // wave m-strip
  const int wn = (wave >> 2) * 32;  // wave n-strip
  const int z  = blockIdx.z;
  const int m0 = blockIdx.y * 128;
  const int n0 = blockIdx.x * 64;
  const bf16_t* Ab = p.A + (long long)z * p.batchA + (long long)m0 * p.lda;
  const bf16_t* Bb = p.B + (long long)z * p.batchB + (long long)n0 * p.ldb;

  // staging coords: A 128 rows x 32k, 2 threads/row (2 async b128 each);
  //                 B  64 rows x 32k, 4 threads/row (1 async b128 each)
  const int ra = tid >> 1, ca = (tid & 1) * 16;
  const int rb = tid >> 2, cb = (tid & 3) * 8;
  const uint32_t ldsA[2] = { (uint32_t)(uintptr_t)&As[0][ra][ca],
                             (uint32_t)(uintptr_t)&As[1][ra][ca] };
  const uint32_t ldsB[2] = { (uint32_t)(uintptr_t)&Bs[0][rb][cb],
                             (uint32_t)(uintptr_t)&Bs[1][rb][cb] };

  v8f acc[2][2] = {};

  auto stage = [&](int buf, int k0){
    const bf16_t* ga = Ab + (long long)ra * p.lda + (k0 + ca);
    const bf16_t* gb = Bb + (long long)rb * p.ldb + (k0 + cb);
    // INST_OFFSET applies to both LDS and global address (ISA 15.18), so the
    // second 16B chunk of the A row uses offset:16 on the same base regs.
    asm volatile("global_load_async_to_lds_b128 %0, %1, off"
                 :: "v"(ldsA[buf]), "v"(ga) : "memory");
    asm volatile("global_load_async_to_lds_b128 %0, %1, off offset:16"
                 :: "v"(ldsA[buf]), "v"(ga) : "memory");
    asm volatile("global_load_async_to_lds_b128 %0, %1, off"
                 :: "v"(ldsB[buf]), "v"(gb) : "memory");
  };

  const int nsteps = p.K >> 5;
  stage(0, 0);
  for (int s = 0; s < nsteps; ++s){
    const int cur = s & 1;
    if (s + 1 < nsteps){
      stage(cur ^ 1, (s + 1) << 5);
      // async loads complete in order: <=3 outstanding retires current tile
      asm volatile("s_wait_asynccnt 3" ::: "memory");
    } else {
      asm volatile("s_wait_asynccnt 0" ::: "memory");
    }
    __syncthreads();

    union frag { v16bf v; uint4 q[2]; };
    frag a[2], b[2];
    #pragma unroll
    for (int mi = 0; mi < 2; ++mi){
      // A layout: lane row m=l16; elems 0-7 -> k=half*8.., elems 8-15 -> k=16+half*8..
      a[mi].q[0] = *(const uint4*)&As[cur][wm + mi*16 + l16][half*8];
      a[mi].q[1] = *(const uint4*)&As[cur][wm + mi*16 + l16][16 + half*8];
    }
    #pragma unroll
    for (int nj = 0; nj < 2; ++nj){
      // B layout: lane col n=l16; elems 0-7 -> k=half*8.., elems 8-15 -> k=16+half*8..
      b[nj].q[0] = *(const uint4*)&Bs[cur][wn + nj*16 + l16][half*8];
      b[nj].q[1] = *(const uint4*)&Bs[cur][wn + nj*16 + l16][16 + half*8];
    }
    #pragma unroll
    for (int mi = 0; mi < 2; ++mi)
      #pragma unroll
      for (int nj = 0; nj < 2; ++nj)
        acc[mi][nj] = __builtin_amdgcn_wmma_f32_16x16x32_bf16(
            false, a[mi].v, false, b[nj].v, (short)0, acc[mi][nj], false, false);
    __syncthreads();
  }

  #pragma unroll
  for (int mi = 0; mi < 2; ++mi){
    #pragma unroll
    for (int nj = 0; nj < 2; ++nj){
      const int n = n0 + wn + nj*16 + l16;
      #pragma unroll
      for (int r = 0; r < 8; ++r){
        const int m = m0 + wm + mi*16 + half*8 + r;
        float v = acc[mi][nj][r];
        switch (p.epilog) {
          case E_PLAIN_F: {
            if (p.bias) v += p.bias[m];
            p.outF[(long long)z*p.batchOut + (long long)m*p.ldout + n] = v;
          } break;
          case E_PLAIN_H: {
            v += p.bias[m];
            p.outH[(long long)z*p.batchOut + (long long)m*p.ldout + n] = f2bf(v);
          } break;
          case E_QSCAT: {          // m=c, n=t, z=b  ->  [b][h][t][d] f32
            v += p.bias[m];
            int hh = m >> 6, d = m & 63;
            p.outF[(((long long)z*H_ + hh)*TT_ + n)*DH_ + d] = v;
          } break;
          case E_SCORES: {         // z=h, m=query-1024, n=key; mask preoffset by b
            float cm = p.mask[TX_ + m] * p.mask[n];
            v = (cm != 0.f) ? v * p.scale : -10000.f;
            p.outF[(long long)z*p.batchOut + (long long)m*p.ldout + n] = v;
          } break;
          case E_PVSCAT: {         // z=h, m=t-1024, n=d -> aoT[t][h*64+d] (base preoffset by b)
            p.outH[(long long)m*C_ + z*DH_ + n] = f2bf(v);
          } break;
          case E_OPROJ: {
            v += p.bias[m];
            v += p.resid[(long long)z*p.batchRes + (long long)m*p.ldres + n];
            p.outF[(long long)z*p.batchOut + (long long)m*p.ldout + n] = v;
          } break;
          case E_GELU_T: {         // transposed write: y1T[t][fc]
            v += p.bias[m];
            float x3 = v*v*v;
            v = 0.5f*v*(1.f + tanhf(0.7978845608f*(v + 0.044715f*x3)));
            v *= p.mask[(long long)z*p.batchMask + n];
            p.outH[(long long)z*p.batchOut + (long long)n*p.ldout + m] = f2bf(v);
          } break;
          case E_CONV2: {
            v += p.bias[m];
            v *= p.mask[(long long)z*p.batchMask + n];
            v += p.resid[(long long)z*p.batchRes + (long long)m*p.ldres + n];
            p.outF[(long long)z*p.batchOut + (long long)m*p.ldout + n] = v;
          } break;
        }
      }
    }
  }
}

// ------------------------------------------------------------- LayerNorm ---
// One wave per (b,t) column over C_=512 channels.
// outH written transposed: outH[(b*TrowsH + colOff + t)*C_ + c]
// outF written plain:      outF[(b*C_ + c)*Tcols + t]
__launch_bounds__(256)
__global__ void layernorm_ch(const float* __restrict__ x, const float* __restrict__ g,
                             const float* __restrict__ bta,
                             bf16_t* outH, int TrowsH, int colOff,
                             float* outF, const float* mask, int Tcols){
  const int wave = threadIdx.x >> 5, lane = threadIdx.x & 31;
  const long long col = (long long)blockIdx.x * 8 + wave;
  const int bi = (int)(col / Tcols), t = (int)(col % Tcols);
  const float* xp = x + ((long long)bi * C_) * Tcols + t;
  float vals[16];
  float s = 0.f;
  #pragma unroll
  for (int i = 0; i < 16; ++i){ vals[i] = xp[(long long)(lane + 32*i) * Tcols]; s += vals[i]; }
  #pragma unroll
  for (int m = 16; m >= 1; m >>= 1) s += __shfl_xor(s, m);
  const float mean = s * (1.f / C_);
  float vs = 0.f;
  #pragma unroll
  for (int i = 0; i < 16; ++i){ float d = vals[i] - mean; vs += d*d; }
  #pragma unroll
  for (int m = 16; m >= 1; m >>= 1) vs += __shfl_xor(vs, m);
  const float rstd = rsqrtf(vs * (1.f / C_) + 1e-5f);
  const float mk = mask ? mask[(long long)bi*Tcols + t] : 1.f;
  #pragma unroll
  for (int i = 0; i < 16; ++i){
    int c = lane + 32*i;
    float o = ((vals[i] - mean) * rstd * g[c] + bta[c]) * mk;
    if (outH) outH[((long long)bi*TrowsH + colOff + t)*C_ + c] = f2bf(o);
    if (outF) outF[((long long)bi*C_ + c)*Tcols + t] = o;
  }
}

// --------------------------------------------------------------- Softmax ---
__launch_bounds__(256)
__global__ void softmax_rows(const float* __restrict__ S, bf16_t* __restrict__ P){
  const long long row = blockIdx.x;
  const float* r = S + row * TT_;
  bf16_t* pr = P + row * TT_;
  const int tid = threadIdx.x;
  float v[8];
  float mx = -3.0e38f;
  #pragma unroll
  for (int i = 0; i < 8; ++i){ v[i] = r[tid + i*256]; mx = fmaxf(mx, v[i]); }
  __shared__ float red[8];
  __shared__ float red2[8];
  #pragma unroll
  for (int m = 16; m >= 1; m >>= 1) mx = fmaxf(mx, __shfl_xor(mx, m));
  if ((tid & 31) == 0) red[tid >> 5] = mx;
  __syncthreads();
  mx = red[0];
  #pragma unroll
  for (int w = 1; w < 8; ++w) mx = fmaxf(mx, red[w]);
  float s = 0.f;
  #pragma unroll
  for (int i = 0; i < 8; ++i){ v[i] = __expf(v[i] - mx); s += v[i]; }
  #pragma unroll
  for (int m = 16; m >= 1; m >>= 1) s += __shfl_xor(s, m);
  if ((tid & 31) == 0) red2[tid >> 5] = s;
  __syncthreads();
  s = 0.f;
  #pragma unroll
  for (int w = 0; w < 8; ++w) s += red2[w];
  const float inv = 1.f / s;
  #pragma unroll
  for (int i = 0; i < 8; ++i) pr[tid + i*256] = f2bf(v[i] * inv);
}

// ------------------------------------------------------------------ RoPE ---
// layout [B,H,T,DH] (f32 in, bf16 out); serves both Q and K.
#define ROPE_LN 0.28782313662425572f  // ln(10000)/32
__global__ void rope_kern(const float* __restrict__ q, bf16_t* __restrict__ o, long long n){
  long long i = (long long)blockIdx.x * 256 + threadIdx.x;
  if (i >= n) return;
  int j = (int)(i & 31);
  long long rest = i >> 5;
  int t = (int)(rest & (TT_ - 1));
  long long z = rest >> 11;
  int pos = t & (TX_ - 1);
  float ang = (float)pos * __expf(-(float)j * ROPE_LN);
  float sn, cs; __sincosf(ang, &sn, &cs);
  long long base = (z * TT_ + t) * DH_ + j;
  float x1 = q[base], x2 = q[base + 32];
  o[base]      = f2bf(x1 * cs - x2 * sn);
  o[base + 32] = f2bf(x2 * cs + x1 * sn);
}

// ---------------------------------------------------------------- im2col ---
// transposed: in  = inT[(b*TX + t')*Cin + c]        (t-major, c contiguous)
//             out = outT[(b*TX + t)*(3*Cin) + c*3+kk] == out[i]
__global__ void im2col3t(const bf16_t* __restrict__ in, const float* __restrict__ mask,
                         bf16_t* __restrict__ out, int Cin, long long n){
  long long i = (long long)blockIdx.x * 256 + threadIdx.x;
  if (i >= n) return;
  const int ck3 = 3 * Cin;
  int ck = (int)(i % ck3);
  long long r = i / ck3;
  int t = (int)(r % TX_);
  int b = (int)(r / TX_);
  int c = ck / 3, kk = ck % 3;
  int ts = t + kk - 1;
  float v = 0.f;
  if (ts >= 0 && ts < TX_){
    v = bf2f(in[((long long)b*TX_ + ts)*Cin + c]);
    if (mask) v *= mask[(long long)b*TX_ + ts];
  }
  out[i] = f2bf(v);
}

// ----------------------------------------------------------- small utils ---
__global__ void cvt_f32_bf16(const float* __restrict__ in, bf16_t* __restrict__ out, long long n){
  long long i = (long long)blockIdx.x * 256 + threadIdx.x;
  if (i < n) out[i] = f2bf(in[i]);
}
__global__ void init_x_kern(const float* __restrict__ x, const float* __restrict__ m,
                            float* __restrict__ o, long long n){
  long long i = (long long)blockIdx.x * 256 + threadIdx.x;
  if (i >= n) return;
  int t = (int)(i % TX_);
  int b = (int)(i / ((long long)C_ * TX_));
  o[i] = x[i] * m[(long long)b*TX_ + t];
}
// h [B,C,TX] -> xcT[(b*TT + t)*C + c]   (left half of concat, transposed)
__global__ void copy_h_xct(const float* __restrict__ h, bf16_t* __restrict__ xcT, long long n){
  long long i = (long long)blockIdx.x * 256 + threadIdx.x;
  if (i >= n) return;
  int t = (int)(i % TX_);
  long long row = i / TX_;            // b*C + c
  int c = (int)(row % C_);
  int b = (int)(row / C_);
  xcT[((long long)b*TT_ + t)*C_ + c] = f2bf(h[i]);
}
__global__ void build_cond(const float* __restrict__ hm, const float* __restrict__ xm,
                           float* __restrict__ cond, long long n){
  long long i = (long long)blockIdx.x * 256 + threadIdx.x;
  if (i >= n) return;
  int s = (int)(i % TT_);
  int b = (int)(i / TT_);
  cond[i] = (s < TX_) ? hm[(long long)b*TX_ + s] : xm[(long long)b*TX_ + (s - TX_)];
}

// ------------------------------------------------------------------ host ---
extern "C" void kernel_launch(void* const* d_in, const int* in_sizes, int n_in,
                              void* d_out, int out_size, void* d_ws, size_t ws_size,
                              hipStream_t stream) {
  (void)in_sizes; (void)n_in; (void)out_size; (void)ws_size;
  const float* x_in  = (const float*)d_in[0];
  const float* xmask = (const float*)d_in[1];
  const float* h_in  = (const float*)d_in[2];
  const float* hmask = (const float*)d_in[3];
  const float* wq = (const float*)d_in[4];   const float* bq = (const float*)d_in[5];
  const float* wk = (const float*)d_in[6];   const float* bk = (const float*)d_in[7];
  const float* wv = (const float*)d_in[8];   const float* bv = (const float*)d_in[9];
  const float* wo = (const float*)d_in[10];  const float* bo = (const float*)d_in[11];
  const float* n1g = (const float*)d_in[12]; const float* n1b = (const float*)d_in[13];
  const float* fw1 = (const float*)d_in[14]; const float* fb1 = (const float*)d_in[15];
  const float* fw2 = (const float*)d_in[16]; const float* fb2 = (const float*)d_in[17];
  const float* n2g = (const float*)d_in[18]; const float* n2b = (const float*)d_in[19];
  const float* fng = (const float*)d_in[20]; const float* fnb = (const float*)d_in[21];
  float* out = (float*)d_out;

  char* ws = (char*)d_ws;
  size_t off = 0;
  auto carve = [&](size_t bytes)->char*{
    char* pp = ws + off; off += (bytes + 255) & ~(size_t)255; return pp;
  };

  float*  cond  = (float*) carve(sizeof(float)  * B_ * TT_);
  float*  xf    = (float*) carve(sizeof(float)  * B_ * C_ * TX_);
  bf16_t* xcT   = (bf16_t*)carve(sizeof(bf16_t) * B_ * TT_ * C_);
  bf16_t* wq_b  = (bf16_t*)carve(sizeof(bf16_t) * C_ * C_);
  bf16_t* wk_b  = (bf16_t*)carve(sizeof(bf16_t) * C_ * C_);
  bf16_t* wv_b  = (bf16_t*)carve(sizeof(bf16_t) * C_ * C_);
  bf16_t* wo_b  = (bf16_t*)carve(sizeof(bf16_t) * C_ * C_);
  bf16_t* w1_b  = (bf16_t*)carve(sizeof(bf16_t) * FC_ * C_ * 3);
  bf16_t* w2_b  = (bf16_t*)carve(sizeof(bf16_t) * C_ * FC_ * 3);
  float*  q_f   = (float*) carve(sizeof(float)  * B_ * C_ * TT_);
  float*  k_f   = (float*) carve(sizeof(float)  * B_ * C_ * TT_);
  bf16_t* q_b   = (bf16_t*)carve(sizeof(bf16_t) * B_ * C_ * TT_);
  bf16_t* k_b   = (bf16_t*)carve(sizeof(bf16_t) * B_ * C_ * TT_);
  bf16_t* v_b   = (bf16_t*)carve(sizeof(bf16_t) * B_ * C_ * TT_);   // plain [C,T] == Vt per head
  float*  scr   = (float*) carve(sizeof(float)  * H_ * TX_ * TT_);  // per-b
  bf16_t* p_b   = (bf16_t*)carve(sizeof(bf16_t) * H_ * TX_ * TT_);  // per-b
  bf16_t* aoT   = (bf16_t*)carve(sizeof(bf16_t) * B_ * TX_ * C_);
  bf16_t* ln2T  = (bf16_t*)carve(sizeof(bf16_t) * B_ * TX_ * C_);
  bf16_t* ic1T  = (bf16_t*)carve(sizeof(bf16_t) * B_ * TX_ * C_ * 3);
  bf16_t* y1T   = (bf16_t*)carve(sizeof(bf16_t) * B_ * TX_ * FC_);
  bf16_t* ic2T  = (bf16_t*)carve(sizeof(bf16_t) * B_ * TX_ * FC_ * 3);

  auto gemm = [&](const GemmP& p, int M, int N, int Z){
    dim3 grd(N / 64, M / 128, Z);
    gemm_bf16_wmma<<<grd, 256, 0, stream>>>(p);
  };
  auto blocks = [](long long n){ return (unsigned)((n + 255) / 256); };

  // ---- prologue -----------------------------------------------------------
  { long long n = (long long)B_*TT_;      build_cond <<<blocks(n),256,0,stream>>>(hmask, xmask, cond, n); }
  { long long n = (long long)B_*C_*TX_;   init_x_kern<<<blocks(n),256,0,stream>>>(x_in, xmask, xf, n); }
  { long long n = (long long)B_*C_*TX_;   copy_h_xct <<<blocks(n),256,0,stream>>>(h_in, xcT, n); }

  const long long NWQ = (long long)C_ * C_;
  const long long NW1 = (long long)FC_ * C_ * 3;
  const long long NW2 = (long long)C_ * FC_ * 3;

  for (int i = 0; i < L_; ++i){
    cvt_f32_bf16<<<blocks(NWQ),256,0,stream>>>(wq + i*NWQ, wq_b, NWQ);
    cvt_f32_bf16<<<blocks(NWQ),256,0,stream>>>(wk + i*NWQ, wk_b, NWQ);
    cvt_f32_bf16<<<blocks(NWQ),256,0,stream>>>(wv + i*NWQ, wv_b, NWQ);
    cvt_f32_bf16<<<blocks(NWQ),256,0,stream>>>(wo + i*NWQ, wo_b, NWQ);
    cvt_f32_bf16<<<blocks(NW1),256,0,stream>>>(fw1 + i*NW1, w1_b, NW1);
    cvt_f32_bf16<<<blocks(NW2),256,0,stream>>>(fw2 + i*NW2, w2_b, NW2);

    // LN1 -> transposed right half of xcT
    layernorm_ch<<<dim3(B_*TX_/8),256,0,stream>>>(xf, n1g + i*C_, n1b + i*C_,
                                                  xcT, TT_, TX_, nullptr, nullptr, TX_);

    // QKV projections: B operand = xcT[(b*TT + t)*C + c]
    { GemmP p{}; p.A=wq_b; p.batchA=0; p.lda=C_;
      p.B=xcT; p.batchB=(long long)TT_*C_; p.ldb=C_; p.K=C_;
      p.bias=bq + i*C_; p.outF=q_f; p.epilog=E_QSCAT; gemm(p, C_, TT_, B_); }
    { GemmP p{}; p.A=wk_b; p.batchA=0; p.lda=C_;
      p.B=xcT; p.batchB=(long long)TT_*C_; p.ldb=C_; p.K=C_;
      p.bias=bk + i*C_; p.outF=k_f; p.epilog=E_QSCAT; gemm(p, C_, TT_, B_); }
    { GemmP p{}; p.A=wv_b; p.batchA=0; p.lda=C_;
      p.B=xcT; p.batchB=(long long)TT_*C_; p.ldb=C_; p.K=C_;
      p.bias=bv + i*C_; p.outH=v_b; p.batchOut=(long long)C_*TT_; p.ldout=TT_;
      p.epilog=E_PLAIN_H; gemm(p, C_, TT_, B_); }

    // RoPE (same [B,H,T,DH] layout for Q and K)
    { long long n = (long long)B_*H_*TT_*32;
      rope_kern<<<blocks(n),256,0,stream>>>(q_f, q_b, n);
      rope_kern<<<blocks(n),256,0,stream>>>(k_f, k_b, n); }

    // attention core per batch (z = head)
    for (int b = 0; b < B_; ++b){
      { GemmP p{}; p.A = q_b + ((long long)b*H_*TT_ + TX_)*DH_; p.batchA=(long long)TT_*DH_; p.lda=DH_;
        p.B = k_b + (long long)b*H_*TT_*DH_; p.batchB=(long long)TT_*DH_; p.ldb=DH_; p.K=DH_;
        p.outF=scr; p.batchOut=(long long)TX_*TT_; p.ldout=TT_;
        p.mask=cond + (long long)b*TT_; p.scale=0.125f; p.epilog=E_SCORES;
        gemm(p, TX_, TT_, H_); }
      softmax_rows<<<dim3(H_*TX_),256,0,stream>>>(scr, p_b);
      { GemmP p{}; p.A=p_b; p.batchA=(long long)TX_*TT_; p.lda=TT_;
        p.B = v_b + (long long)b*C_*TT_; p.batchB=(long long)DH_*TT_; p.ldb=TT_; p.K=TT_;
        p.outH = aoT + (long long)b*TX_*C_; p.epilog=E_PVSCAT;
        gemm(p, TX_, DH_, H_); }
    }

    // output projection + residual; B operand = aoT[(b*TX + t)*C + c]
    { GemmP p{}; p.A=wo_b; p.batchA=0; p.lda=C_;
      p.B=aoT; p.batchB=(long long)TX_*C_; p.ldb=C_; p.K=C_;
      p.bias=bo + i*C_;
      p.resid=xf; p.batchRes=(long long)C_*TX_; p.ldres=TX_;
      p.outF=xf; p.batchOut=(long long)C_*TX_; p.ldout=TX_;
      p.epilog=E_OPROJ; gemm(p, C_, TX_, B_); }

    // LN2 -> transposed ln2T
    layernorm_ch<<<dim3(B_*TX_/8),256,0,stream>>>(xf, n2g + i*C_, n2b + i*C_,
                                                  ln2T, TX_, 0, nullptr, nullptr, TX_);

    // FFN conv1
    { long long n = (long long)B_*TX_*C_*3;
      im2col3t<<<blocks(n),256,0,stream>>>(ln2T, xmask, ic1T, C_, n); }
    { GemmP p{}; p.A=w1_b; p.batchA=0; p.lda=C_*3;
      p.B=ic1T; p.batchB=(long long)TX_*C_*3; p.ldb=C_*3; p.K=C_*3;
      p.bias=fb1 + i*FC_;
      p.mask=xmask; p.batchMask=TX_;
      p.outH=y1T; p.batchOut=(long long)TX_*FC_; p.ldout=FC_;
      p.epilog=E_GELU_T; gemm(p, FC_, TX_, B_); }

    // FFN conv2 (+ mask + residual)
    { long long n = (long long)B_*TX_*FC_*3;
      im2col3t<<<blocks(n),256,0,stream>>>(y1T, nullptr, ic2T, FC_, n); }
    { GemmP p{}; p.A=w2_b; p.batchA=0; p.lda=FC_*3;
      p.B=ic2T; p.batchB=(long long)TX_*FC_*3; p.ldb=FC_*3; p.K=FC_*3;
      p.bias=fb2 + i*C_;
      p.mask=xmask; p.batchMask=TX_;
      p.resid=xf; p.batchRes=(long long)C_*TX_; p.ldres=TX_;
      p.outF=xf; p.batchOut=(long long)C_*TX_; p.ldout=TX_;
      p.epilog=E_CONV2; gemm(p, C_, TX_, B_); }
  }

  // final LN * mask -> d_out (fp32, [B,C,Tx])
  layernorm_ch<<<dim3(B_*TX_/8),256,0,stream>>>(xf, fng, fnb,
                                                nullptr, 0, 0, out, xmask, TX_);
}
